// DecoderOnlyExpanderRVQ_678604833531
// MI455X (gfx1250) — compile-verified, matrix operand order checked by reading, not run on hardware
//
#include <hip/hip_runtime.h>
#include <hip/hip_bf16.h>

// ---------------------------------------------------------------------------
// CDNA5 (gfx1250) wave32 WMMA / TDM types
// ---------------------------------------------------------------------------
typedef __attribute__((ext_vector_type(16))) __bf16 v16bf;
typedef __attribute__((ext_vector_type(8)))  float  v8f;
typedef __attribute__((ext_vector_type(4)))  unsigned int u32x4;
typedef __attribute__((ext_vector_type(8)))  int i32x8;
typedef __attribute__((ext_vector_type(4)))  int i32x4;

#if defined(__has_builtin)
#  if __has_builtin(__builtin_amdgcn_tensor_load_to_lds)
#    define HAVE_TDM 1
#  endif
#endif
#ifndef HAVE_TDM
#  define HAVE_TDM 0
#endif

static __device__ __forceinline__ unsigned short f2bf_u(float f) {
    unsigned u = __float_as_uint(f);
    u += 0x7FFFu + ((u >> 16) & 1u);            // round-to-nearest-even
    return (unsigned short)(u >> 16);
}
static __device__ __forceinline__ __bf16 f2bf(float f) {
    unsigned short h = f2bf_u(f);
    return __builtin_bit_cast(__bf16, h);
}
static __device__ __forceinline__ unsigned packbf(float lo, float hi) {
    return (unsigned)f2bf_u(lo) | ((unsigned)f2bf_u(hi) << 16);
}
static __device__ __forceinline__ float bf2f(__bf16 b) {
    unsigned short s = __builtin_bit_cast(unsigned short, b);
    return __uint_as_float(((unsigned)s) << 16);
}

#define NEGF (-1e30f)

// ---------------------------------------------------------------------------
// f32 -> packed bf16 conversion (weights / memory), 4 elems per thread
// ---------------------------------------------------------------------------
__global__ void cvt_f32_bf16_kernel(const float* __restrict__ src,
                                    __bf16* __restrict__ dst, size_t n4)
{
    size_t i = (size_t)blockIdx.x * 256 + threadIdx.x;
    if (i >= n4) return;
    float4 v = *(const float4*)(src + i * 4);
    unsigned* d = (unsigned*)(dst + i * 4);
    d[0] = packbf(v.x, v.y);
    d[1] = packbf(v.z, v.w);
}

// ---------------------------------------------------------------------------
// RMSNorm: one 256-thread block per row, D = 1024 (4 contiguous elems/thread)
// outf != nullptr -> f32 output, else packed bf16 output
// ---------------------------------------------------------------------------
__global__ __launch_bounds__(256) void rmsnorm_kernel(
    const float* __restrict__ x, const float* __restrict__ w,
    float* __restrict__ outf, __bf16* __restrict__ outb, int D)
{
    const int row = blockIdx.x;
    const int tid = threadIdx.x;
    const float* xr = x + (size_t)row * D;
    float4 v = *(const float4*)(xr + tid * 4);
    float ss = v.x * v.x + v.y * v.y + v.z * v.z + v.w * v.w;
    for (int off = 16; off > 0; off >>= 1) ss += __shfl_xor(ss, off, 32);
    __shared__ float red[8];
    if ((tid & 31) == 0) red[tid >> 5] = ss;
    __syncthreads();
    float tot = 0.f;
    for (int i = 0; i < 8; ++i) tot += red[i];
    float sc = rsqrtf(tot / (float)D + 1e-6f);
    float4 wv = *(const float4*)(w + tid * 4);
    float o0 = v.x * sc * wv.x, o1 = v.y * sc * wv.y;
    float o2 = v.z * sc * wv.z, o3 = v.w * sc * wv.w;
    if (outf) {
        *(float4*)(outf + (size_t)row * D + tid * 4) = make_float4(o0, o1, o2, o3);
    } else {
        unsigned* d = (unsigned*)(outb + (size_t)row * D + tid * 4);
        d[0] = packbf(o0, o1);
        d[1] = packbf(o2, o3);
    }
}

// ---------------------------------------------------------------------------
// Tensor Data Mover: load a 128x64 bf16 tile (row stride = dim0_stride elems)
// into LDS with hardware padding of 4 dwords every 32 dwords -> LDS row
// stride of 72 bf16, matching the [128][72] shared tiles below.
// ---------------------------------------------------------------------------
#define BM 128
#define BN 128
#define BK 64
#define LSTR 72

#if HAVE_TDM
static __device__ __forceinline__ void tdm_load_tile(
    unsigned lds_off, const __bf16* gsrc, int dim0, int dim1, int stride0)
{
    unsigned long long ga = (unsigned long long)(size_t)gsrc;
    u32x4 g0;
    g0[0] = 1u;                                            // count=1 (valid user D#)
    g0[1] = lds_off;                                       // lds_addr (bytes)
    g0[2] = (unsigned)ga;                                  // global_addr[31:0]
    g0[3] = (unsigned)((ga >> 32) & 0x01FFFFFFull)         // global_addr[56:32]
          | (2u << 30);                                    // type = 2 (image)
    i32x8 g1;
    g1[0] = (1 << 16)                                      // data_size = 2 bytes
          | (1 << 20)                                      // pad_enable
          | (4 << 22)                                      // pad_interval: 32 dwords
          | (3 << 25);                                     // pad_amount: 4 dwords
    g1[1] = (int)((unsigned)(dim0 & 0xFFFF) << 16);        // tensor_dim0[15:0]
    g1[2] = (int)(((unsigned)dim0 >> 16) | ((unsigned)(dim1 & 0xFFFF) << 16));
    g1[3] = (int)((((unsigned)dim1 >> 16) & 0xFFFFu) | ((unsigned)BK << 16)); // tile_dim0
    g1[4] = BM;                                            // tile_dim1=128, tile_dim2=0
    g1[5] = stride0;                                       // tensor_dim0_stride[31:0]
    g1[6] = 0;
    g1[7] = 0;
    i32x4 z4 = {};
#if __clang_major__ >= 23
    i32x8 z8 = {};
    __builtin_amdgcn_tensor_load_to_lds(g0, g1, z4, z4, z8, 0);
#else
    __builtin_amdgcn_tensor_load_to_lds(g0, g1, z4, z4, 0);
#endif
}
#endif

// ---------------------------------------------------------------------------
// GEMM: C[M,N] = A[M,K] @ W[N,K]^T (+bias) (+res). A, W bf16; C f32 or bf16.
// 128x128x64 tile, 8 waves (2x4). Double-buffered TDM pipeline: stage i+1's
// two tensor_load_to_lds are issued before computing stage i, and the wave
// waits only for TENSORcnt<=2 so the DMA overlaps the WMMA work.
// ---------------------------------------------------------------------------
__global__ __launch_bounds__(256) void gemm_kernel(
    const __bf16* __restrict__ A, const __bf16* __restrict__ W,
    const float* __restrict__ bias, const float* __restrict__ res,
    float* __restrict__ Cf, __bf16* __restrict__ Cb, int M, int N, int K)
{
    __shared__ __bf16 ldsA[2][BM][LSTR];
    __shared__ __bf16 ldsB[2][BN][LSTR];

    const int tid  = threadIdx.x;
    const int wave = tid >> 5, lane = tid & 31;
    const int lm = lane & 15, hf = lane >> 4;
    const int waveRow = wave >> 2;   // 0..1 -> 64 rows
    const int waveCol = wave & 3;    // 0..3 -> 32 cols
    const int rowBase = blockIdx.y * BM;
    const int colBase = blockIdx.x * BN;

    const __bf16* Abase = A + (size_t)rowBase * K;
    const __bf16* Wbase = W + (size_t)colBase * K;
    const int nstage = K / BK;

    v8f acc[4][2] = {};

#if HAVE_TDM
    if (tid < 32) {   // wave 0 primes the pipeline (stage 0 -> buffer 0)
        tdm_load_tile((unsigned)(size_t)&ldsA[0][0][0], Abase, K, M, K);
        tdm_load_tile((unsigned)(size_t)&ldsB[0][0][0], Wbase, K, N, K);
    }
#endif

    for (int st = 0; st < nstage; ++st) {
        const int cur = st & 1;
#if HAVE_TDM
        if (tid < 32) {
            if (st + 1 < nstage) {   // issue next stage into the other buffer
                const int nxt = cur ^ 1;
                tdm_load_tile((unsigned)(size_t)&ldsA[nxt][0][0],
                              Abase + (size_t)(st + 1) * BK, K, M, K);
                tdm_load_tile((unsigned)(size_t)&ldsB[nxt][0][0],
                              Wbase + (size_t)(st + 1) * BK, K, N, K);
                __builtin_amdgcn_s_wait_tensorcnt(2);   // retire current stage's pair
            } else {
                __builtin_amdgcn_s_wait_tensorcnt(0);
            }
        }
#else
        for (int i = tid; i < (BM * BK) / 8; i += 256) {
            int r = i >> 3, c8 = (i & 7) * 8;
            *(uint4*)&ldsA[cur][r][c8] = *(const uint4*)(Abase + (size_t)r * K + st * BK + c8);
            *(uint4*)&ldsB[cur][r][c8] = *(const uint4*)(Wbase + (size_t)r * K + st * BK + c8);
        }
#endif
        __syncthreads();

        for (int kk = 0; kk < BK; kk += 32) {
            v16bf bfrag[2];
            for (int nt = 0; nt < 2; ++nt) {
                const __bf16* p = &ldsB[cur][waveCol * 32 + nt * 16 + lm][kk + hf * 16];
                for (int t = 0; t < 16; ++t) bfrag[nt][t] = p[t];
            }
            for (int mt = 0; mt < 4; ++mt) {
                v16bf afrag;
                const __bf16* p = &ldsA[cur][waveRow * 64 + mt * 16 + lm][kk + hf * 16];
                for (int t = 0; t < 16; ++t) afrag[t] = p[t];
                for (int nt = 0; nt < 2; ++nt)
                    acc[mt][nt] = __builtin_amdgcn_wmma_f32_16x16x32_bf16(
                        false, afrag, false, bfrag[nt], (short)0, acc[mt][nt], false, false);
            }
        }
        __syncthreads();   // all waves done with buffer `cur` before it is re-filled
    }

    for (int mt = 0; mt < 4; ++mt)
        for (int nt = 0; nt < 2; ++nt)
            for (int j = 0; j < 8; ++j) {
                int row = rowBase + waveRow * 64 + mt * 16 + j + hf * 8;
                int col = colBase + waveCol * 32 + nt * 16 + lm;
                float v = acc[mt][nt][j];
                if (bias) v += bias[col];
                size_t idx = (size_t)row * N + col;
                if (Cf) {
                    if (res) v += res[idx];
                    Cf[idx] = v;
                } else {
                    Cb[idx] = f2bf(v);
                }
            }
}

// ---------------------------------------------------------------------------
// RoPE in-place on q and k inside qkv buffer [B, L, 3D] (f32)
// ---------------------------------------------------------------------------
__global__ void rope_kernel(float* __restrict__ qkv, int B, int L, int D)
{
    int idx = blockIdx.x * 256 + threadIdx.x;     // B*L*H*32 threads
    int d2 = idx & 31;
    int t  = idx >> 5;
    int h  = t & 15;
    t >>= 4;
    int pos = t % L;
    int b   = t / L;
    if (b >= B) return;
    float inv = __powf(10000.f, -(float)d2 / 32.f);
    float ang = (float)pos * inv;
    float cs = __cosf(ang), sn = __sinf(ang);
    float* qb = qkv + (size_t)(b * L + pos) * (3 * D) + h * 64;
    float x1 = qb[d2], x2 = qb[d2 + 32];
    qb[d2]      = x1 * cs - x2 * sn;
    qb[d2 + 32] = x2 * cs + x1 * sn;
    float* kb = qb + D;
    x1 = kb[d2]; x2 = kb[d2 + 32];
    kb[d2]      = x1 * cs - x2 * sn;
    kb[d2 + 32] = x2 * cs + x1 * sn;
}

// ---------------------------------------------------------------------------
// Flash self-attention (causal). grid = (L/64, H, B), 128 threads = 4 waves.
// qkv is f32 (post-RoPE); output is bf16 (feeds Wo GEMM).
// ---------------------------------------------------------------------------
__global__ __launch_bounds__(128) void self_attn_kernel(
    const float* __restrict__ qkv, __bf16* __restrict__ out,
    int B, int L, int D)
{
    const int b = blockIdx.z, hh = blockIdx.y, qt = blockIdx.x;
    const int tid = threadIdx.x;
    const int wave = tid >> 5, lane = tid & 31;
    const int lm = lane & 15, hf = lane >> 4;
    const int qbase = qt * 64;
    const int TD = 3 * D;

    __shared__ __bf16 ldsK [64][72];   // [key][d]
    __shared__ __bf16 ldsVt[64][72];   // [d][key]
    __shared__ __bf16 ldsP [64][72];   // [qrow][key], wave-private rows

    v16bf qf[2];
    {
        const float* qp = qkv + (size_t)(b * L + qbase + wave * 16 + lm) * TD + hh * 64;
        for (int c2 = 0; c2 < 2; ++c2)
            for (int t4 = 0; t4 < 16; t4 += 4) {
                float4 q4 = *(const float4*)(qp + c2 * 32 + hf * 16 + t4);
                qf[c2][t4 + 0] = f2bf(q4.x * 0.125f);
                qf[c2][t4 + 1] = f2bf(q4.y * 0.125f);
                qf[c2][t4 + 2] = f2bf(q4.z * 0.125f);
                qf[c2][t4 + 3] = f2bf(q4.w * 0.125f);
            }
    }

    float m_i[8], l_i[8];
    v8f o_acc[4] = {};
    for (int j = 0; j < 8; ++j) { m_i[j] = -3.0e38f; l_i[j] = 0.f; }

    const int r = tid >> 1, half = (tid & 1) * 32;

    for (int kt = 0; kt <= qt; ++kt) {
        const int kb = kt * 64;
        {
            const float* kp = qkv + (size_t)(b * L + kb + r) * TD + D + hh * 64 + half;
            const float* vp = kp + D;
            for (int c = 0; c < 32; c += 4) {
                float4 k4 = *(const float4*)(kp + c);
                unsigned* kd = (unsigned*)&ldsK[r][half + c];
                kd[0] = packbf(k4.x, k4.y);
                kd[1] = packbf(k4.z, k4.w);
                float4 v4 = *(const float4*)(vp + c);
                ldsVt[half + c + 0][r] = f2bf(v4.x);
                ldsVt[half + c + 1][r] = f2bf(v4.y);
                ldsVt[half + c + 2][r] = f2bf(v4.z);
                ldsVt[half + c + 3][r] = f2bf(v4.w);
            }
        }
        __syncthreads();

        v8f s[4] = {};
        for (int c2 = 0; c2 < 2; ++c2)
            for (int n = 0; n < 4; ++n) {
                v16bf kf;
                const __bf16* p = &ldsK[n * 16 + lm][c2 * 32 + hf * 16];
                for (int t = 0; t < 16; ++t) kf[t] = p[t];
                s[n] = __builtin_amdgcn_wmma_f32_16x16x32_bf16(
                    false, qf[c2], false, kf, (short)0, s[n], false, false);
            }

        if (kt == qt) {
            for (int n = 0; n < 4; ++n)
                for (int j = 0; j < 8; ++j) {
                    int row = qbase + wave * 16 + j + hf * 8;
                    int col = kb + n * 16 + lm;
                    if (col > row) s[n][j] = NEGF;
                }
        }

        for (int j = 0; j < 8; ++j) {
            float mx = s[0][j];
            for (int n = 1; n < 4; ++n) mx = fmaxf(mx, s[n][j]);
            for (int off = 1; off < 16; off <<= 1) mx = fmaxf(mx, __shfl_xor(mx, off, 32));
            float mnew = fmaxf(m_i[j], mx);
            float corr = __expf(m_i[j] - mnew);
            m_i[j] = mnew;
            float rs = 0.f;
            for (int n = 0; n < 4; ++n) {
                float p = __expf(s[n][j] - mnew);
                s[n][j] = p;
                rs += p;
            }
            for (int off = 1; off < 16; off <<= 1) rs += __shfl_xor(rs, off, 32);
            l_i[j] = l_i[j] * corr + rs;
            for (int n = 0; n < 4; ++n) o_acc[n][j] *= corr;
        }

        for (int n = 0; n < 4; ++n)
            for (int j = 0; j < 8; ++j)
                ldsP[wave * 16 + j + hf * 8][n * 16 + lm] = f2bf(s[n][j]);

        for (int c2 = 0; c2 < 2; ++c2) {
            v16bf pa;
            const __bf16* pp = &ldsP[wave * 16 + lm][c2 * 32 + hf * 16];
            for (int t = 0; t < 16; ++t) pa[t] = pp[t];
            for (int n = 0; n < 4; ++n) {
                v16bf vf;
                const __bf16* vp2 = &ldsVt[n * 16 + lm][c2 * 32 + hf * 16];
                for (int t = 0; t < 16; ++t) vf[t] = vp2[t];
                o_acc[n] = __builtin_amdgcn_wmma_f32_16x16x32_bf16(
                    false, pa, false, vf, (short)0, o_acc[n], false, false);
            }
        }
        __syncthreads();
    }

    for (int n = 0; n < 4; ++n)
        for (int j = 0; j < 8; ++j) {
            int row = qbase + wave * 16 + j + hf * 8;
            int col = n * 16 + lm;
            out[(size_t)(b * L + row) * D + hh * 64 + col] = f2bf(o_acc[n][j] / l_i[j]);
        }
}

// ---------------------------------------------------------------------------
// Flash cross-attention with segment lookback mask. q, kv are bf16 already.
// q: [B,L,D], kv: [B,S,2D]. grid = (L/64, H, B), 128 threads. out bf16.
// ---------------------------------------------------------------------------
__global__ __launch_bounds__(128) void cross_attn_kernel(
    const __bf16* __restrict__ q, const __bf16* __restrict__ kv,
    const int* __restrict__ seg, __bf16* __restrict__ out,
    int B, int L, int S, int D)
{
    const int b = blockIdx.z, hh = blockIdx.y, qt = blockIdx.x;
    const int tid = threadIdx.x;
    const int wave = tid >> 5, lane = tid & 31;
    const int lm = lane & 15, hf = lane >> 4;
    const int qbase = qt * 64;

    __shared__ __bf16 ldsK [64][72];
    __shared__ __bf16 ldsVt[64][72];
    __shared__ __bf16 ldsP [64][72];

    v16bf qf[2];
    {
        const __bf16* qp = q + (size_t)(b * L + qbase + wave * 16 + lm) * D + hh * 64;
        for (int c2 = 0; c2 < 2; ++c2)
            for (int t = 0; t < 16; ++t)
                qf[c2][t] = f2bf(bf2f(qp[c2 * 32 + hf * 16 + t]) * 0.125f);
    }
    int seg_j[8];
    for (int j = 0; j < 8; ++j)
        seg_j[j] = seg[(size_t)b * L + qbase + wave * 16 + j + hf * 8];

    float m_i[8], l_i[8];
    v8f o_acc[4] = {};
    for (int j = 0; j < 8; ++j) { m_i[j] = -3.0e38f; l_i[j] = 0.f; }

    const int r = tid >> 1, half = (tid & 1) * 32;

    for (int kt = 0; kt < S / 64; ++kt) {
        const int kb = kt * 64;
        {
            const __bf16* kp = kv + (size_t)(b * S + kb + r) * (2 * D) + hh * 64 + half;
            const __bf16* vp = kp + D;
            *(uint4*)&ldsK[r][half +  0] = *(const uint4*)(kp);
            *(uint4*)&ldsK[r][half +  8] = *(const uint4*)(kp + 8);
            *(uint4*)&ldsK[r][half + 16] = *(const uint4*)(kp + 16);
            *(uint4*)&ldsK[r][half + 24] = *(const uint4*)(kp + 24);
            for (int c = 0; c < 32; ++c) ldsVt[half + c][r] = vp[c];
        }
        __syncthreads();

        v8f s[4] = {};
        for (int c2 = 0; c2 < 2; ++c2)
            for (int n = 0; n < 4; ++n) {
                v16bf kf;
                const __bf16* p = &ldsK[n * 16 + lm][c2 * 32 + hf * 16];
                for (int t = 0; t < 16; ++t) kf[t] = p[t];
                s[n] = __builtin_amdgcn_wmma_f32_16x16x32_bf16(
                    false, qf[c2], false, kf, (short)0, s[n], false, false);
            }

        for (int n = 0; n < 4; ++n)
            for (int j = 0; j < 8; ++j) {
                int col = kb + n * 16 + lm;
                int sg = seg_j[j];
                if (!(col <= sg && col > sg - 128)) s[n][j] = NEGF;
            }

        for (int j = 0; j < 8; ++j) {
            float mx = s[0][j];
            for (int n = 1; n < 4; ++n) mx = fmaxf(mx, s[n][j]);
            for (int off = 1; off < 16; off <<= 1) mx = fmaxf(mx, __shfl_xor(mx, off, 32));
            float mnew = fmaxf(m_i[j], mx);
            float corr = __expf(m_i[j] - mnew);
            m_i[j] = mnew;
            float rs = 0.f;
            for (int n = 0; n < 4; ++n) {
                float p = __expf(s[n][j] - mnew);
                s[n][j] = p;
                rs += p;
            }
            for (int off = 1; off < 16; off <<= 1) rs += __shfl_xor(rs, off, 32);
            l_i[j] = l_i[j] * corr + rs;
            for (int n = 0; n < 4; ++n) o_acc[n][j] *= corr;
        }

        for (int n = 0; n < 4; ++n)
            for (int j = 0; j < 8; ++j)
                ldsP[wave * 16 + j + hf * 8][n * 16 + lm] = f2bf(s[n][j]);

        for (int c2 = 0; c2 < 2; ++c2) {
            v16bf pa;
            const __bf16* pp = &ldsP[wave * 16 + lm][c2 * 32 + hf * 16];
            for (int t = 0; t < 16; ++t) pa[t] = pp[t];
            for (int n = 0; n < 4; ++n) {
                v16bf vf;
                const __bf16* vp2 = &ldsVt[n * 16 + lm][c2 * 32 + hf * 16];
                for (int t = 0; t < 16; ++t) vf[t] = vp2[t];
                o_acc[n] = __builtin_amdgcn_wmma_f32_16x16x32_bf16(
                    false, pa, false, vf, (short)0, o_acc[n], false, false);
            }
        }
        __syncthreads();
    }

    for (int n = 0; n < 4; ++n)
        for (int j = 0; j < 8; ++j) {
            int row = qbase + wave * 16 + j + hf * 8;
            int col = n * 16 + lm;
            out[(size_t)(b * L + row) * D + hh * 64 + col] = f2bf(o_acc[n][j] / l_i[j]);
        }
}

// ---------------------------------------------------------------------------
// SwiGLU elementwise on bf16: g = silu(g) * u, 4 elems/thread
// ---------------------------------------------------------------------------
__global__ void silu_mul_kernel(__bf16* __restrict__ g, const __bf16* __restrict__ u,
                                size_t n4)
{
    size_t i = (size_t)blockIdx.x * 256 + threadIdx.x;
    if (i >= n4) return;
    __bf16* gp = g + i * 4;
    const __bf16* up = u + i * 4;
    float o[4];
    for (int k = 0; k < 4; ++k) {
        float t = bf2f(gp[k]);
        o[k] = (t / (1.f + __expf(-t))) * bf2f(up[k]);
    }
    unsigned* d = (unsigned*)gp;
    d[0] = packbf(o[0], o[1]);
    d[1] = packbf(o[2], o[3]);
}

// ---------------------------------------------------------------------------
// Host orchestration
// ---------------------------------------------------------------------------
static void launch_cvt(const float* src, __bf16* dst, size_t n, hipStream_t s)
{
    size_t n4 = n / 4;
    cvt_f32_bf16_kernel<<<(unsigned)((n4 + 255) / 256), 256, 0, s>>>(src, dst, n4);
}

static void launch_gemm(const __bf16* A, const __bf16* W, const float* bias,
                        const float* res, float* Cf, __bf16* Cb,
                        int M, int N, int K, hipStream_t s)
{
    dim3 g(N / BN, M / BM);
    gemm_kernel<<<g, 256, 0, s>>>(A, W, bias, res, Cf, Cb, M, N, K);
}

extern "C" void kernel_launch(void* const* d_in, const int* in_sizes, int n_in,
                              void* d_out, int out_size, void* d_ws, size_t ws_size,
                              hipStream_t stream)
{
    (void)in_sizes; (void)n_in; (void)out_size;
    const int B = 2, L = 2048, S = 256, D = 1024, Hn = 16, F = 4096;
    const int NL = 4;

    const float* x_in   = (const float*)d_in[0];
    const float* memory = (const float*)d_in[1];
    const int*   seg    = (const int*)  d_in[2];
    const float* Wqkv   = (const float*)d_in[3];
    const float* bqkv   = (const float*)d_in[4];
    const float* Wo     = (const float*)d_in[5];
    const float* bo     = (const float*)d_in[6];
    const float* Wq_c   = (const float*)d_in[7];
    const float* Wkv_c  = (const float*)d_in[8];
    const float* Wo_c   = (const float*)d_in[9];
    const float* Wg     = (const float*)d_in[10];
    const float* Wu     = (const float*)d_in[11];
    const float* Wd     = (const float*)d_in[12];
    const float* n1     = (const float*)d_in[13];
    const float* n2     = (const float*)d_in[14];
    const float* n3     = (const float*)d_in[15];
    const float* nf     = (const float*)d_in[16];

    float* x = (float*)d_out;            // residual stream lives in d_out (f32)

    // ---- workspace carve-up (bytes) ----
    char* p = (char*)d_ws;
    const size_t ROWS = (size_t)B * L;                 // 4096
    __bf16* wbf    = (__bf16*)p; p += (size_t)F * D * sizeof(__bf16);        // weight scratch
    __bf16* hbuf   = (__bf16*)p; p += ROWS * D * sizeof(__bf16);             // normed acts
    float*  qkvbuf = (float*) p; p += ROWS * 3 * D * sizeof(float);          // f32 (RoPE)
    __bf16* abuf   = (__bf16*)p; p += ROWS * D * sizeof(__bf16);             // attn outs
    __bf16* qcbuf  = (__bf16*)p; p += ROWS * D * sizeof(__bf16);             // cross Q
    __bf16* kvbuf  = (__bf16*)p; p += (size_t)B * S * 2 * D * sizeof(__bf16);
    __bf16* membf  = (__bf16*)p; p += (size_t)B * S * D * sizeof(__bf16);
    __bf16* gbuf   = (__bf16*)p; p += ROWS * F * sizeof(__bf16);
    __bf16* ubuf   = (__bf16*)p; p += ROWS * F * sizeof(__bf16);
    if ((size_t)(p - (char*)d_ws) > ws_size) return;

    hipMemcpyAsync(x, x_in, ROWS * D * sizeof(float), hipMemcpyDeviceToDevice, stream);
    launch_cvt(memory, membf, (size_t)B * S * D, stream);

    const dim3 attnGrid(L / 64, Hn, B);
    const int M = (int)ROWS;

    for (int l = 0; l < NL; ++l) {
        const float* Wqkv_l = Wqkv + (size_t)l * 3 * D * D;
        const float* bqkv_l = bqkv + (size_t)l * 3 * D;
        const float* Wo_l   = Wo   + (size_t)l * D * D;
        const float* bo_l   = bo   + (size_t)l * D;
        const float* Wq_c_l = Wq_c + (size_t)l * D * D;
        const float* Wkv_l  = Wkv_c+ (size_t)l * 2 * D * D;
        const float* Wo_c_l = Wo_c + (size_t)l * D * D;
        const float* Wg_l   = Wg   + (size_t)l * F * D;
        const float* Wu_l   = Wu   + (size_t)l * F * D;
        const float* Wd_l   = Wd   + (size_t)l * D * F;

        // --- self-attention block ---
        rmsnorm_kernel<<<M, 256, 0, stream>>>(x, n1 + (size_t)l * D, nullptr, hbuf, D);
        launch_cvt(Wqkv_l, wbf, (size_t)3 * D * D, stream);
        launch_gemm(hbuf, wbf, bqkv_l, nullptr, qkvbuf, nullptr, M, 3 * D, D, stream);
        rope_kernel<<<(B * L * Hn * 32) / 256, 256, 0, stream>>>(qkvbuf, B, L, D);
        self_attn_kernel<<<attnGrid, 128, 0, stream>>>(qkvbuf, abuf, B, L, D);
        launch_cvt(Wo_l, wbf, (size_t)D * D, stream);
        launch_gemm(abuf, wbf, bo_l, x, x, nullptr, M, D, D, stream);

        // --- cross-attention block ---
        rmsnorm_kernel<<<M, 256, 0, stream>>>(x, n2 + (size_t)l * D, nullptr, hbuf, D);
        launch_cvt(Wq_c_l, wbf, (size_t)D * D, stream);
        launch_gemm(hbuf, wbf, nullptr, nullptr, nullptr, qcbuf, M, D, D, stream);
        launch_cvt(Wkv_l, wbf, (size_t)2 * D * D, stream);
        launch_gemm(membf, wbf, nullptr, nullptr, nullptr, kvbuf, B * S, 2 * D, D, stream);
        cross_attn_kernel<<<attnGrid, 128, 0, stream>>>(qcbuf, kvbuf, seg, abuf, B, L, S, D);
        launch_cvt(Wo_c_l, wbf, (size_t)D * D, stream);
        launch_gemm(abuf, wbf, nullptr, x, x, nullptr, M, D, D, stream);

        // --- SwiGLU MLP ---
        rmsnorm_kernel<<<M, 256, 0, stream>>>(x, n3 + (size_t)l * D, nullptr, hbuf, D);
        launch_cvt(Wg_l, wbf, (size_t)F * D, stream);
        launch_gemm(hbuf, wbf, nullptr, nullptr, nullptr, gbuf, M, F, D, stream);
        launch_cvt(Wu_l, wbf, (size_t)F * D, stream);
        launch_gemm(hbuf, wbf, nullptr, nullptr, nullptr, ubuf, M, F, D, stream);
        silu_mul_kernel<<<(unsigned)((ROWS * F / 4 + 255) / 256), 256, 0, stream>>>(
            gbuf, ubuf, ROWS * F / 4);
        launch_cvt(Wd_l, wbf, (size_t)D * F, stream);
        launch_gemm(gbuf, wbf, nullptr, x, x, nullptr, M, D, F, stream);
    }

    // final norm, f32, in place on d_out
    rmsnorm_kernel<<<M, 256, 0, stream>>>(x, nf, x, nullptr, D);
}
